// ContrastiveLoss_15590731284678
// MI455X (gfx1250) — matrix operand order repaired and checked by reading, not compiled, not run on hardware
//
#include <hip/hip_runtime.h>

// ---------------- types ----------------
typedef __bf16 bf16_t;
typedef __attribute__((ext_vector_type(16))) __bf16 v16bf;
typedef __attribute__((ext_vector_type(8)))  __bf16 v8bf;
typedef __attribute__((ext_vector_type(4)))  __bf16 v4bf;
typedef __attribute__((ext_vector_type(8)))  float  v8f;
typedef __attribute__((ext_vector_type(4)))  int    v4i;

// ---------------- problem constants ----------------
#define B_ROWS   8192
#define D_DIM    1024
#define INV_TAU  14.2857142857142857f   // 1/0.07
#define EPS_NRM  1e-12f

// ---------------- GEMM tiling ----------------
#define MT      128        // rows per workgroup
#define NT      128        // cols per tile
#define KSTAGE  64         // K elements staged in LDS per pipeline stage
#define NSTAGES (D_DIM / KSTAGE)          // 16
#define NSEG    8          // N-sweep split across blockIdx.y
#define NT_PER_SEG (B_ROWS / NT / NSEG)   // 8 tiles of 128 per segment
#define LDS_STRIDE 72      // bf16 elems per LDS row (64 data + 8 pad) = 144B, 16B-aligned

// ---------------- CDNA5 async global->LDS path (guarded) ----------------
#if defined(__has_builtin)
#if __has_builtin(__builtin_amdgcn_global_load_async_to_lds_b128) && \
    __has_builtin(__builtin_amdgcn_s_wait_asynccnt)
#define USE_ASYNC_LDS 1
#endif
#endif

#ifdef USE_ASYNC_LDS
// builtin expects pointers-to-int4 in AS1 (global) / AS3 (LDS)
#define AS1PTR(p) ((__attribute__((address_space(1))) v4i*)((void*)(p)))
#define AS3PTR(p) ((__attribute__((address_space(3))) v4i*)((void*)(p)))
#endif

// ============================================================
// Kernel 0: zero the per-row exp-sum accumulator
// ============================================================
__global__ void cl_zero_kernel(float* __restrict__ p) {
    p[blockIdx.x * 256 + threadIdx.x] = 0.0f;
}

// ============================================================
// Kernel 1: row L2-normalize + split into bf16 hi/lo planes
// grid: 2*B_ROWS blocks of 128 threads (first half = x, second = y)
// ============================================================
__device__ __forceinline__ void cl_split_store(bf16_t* hi, bf16_t* lo,
                                               float4 v, float s) {
    v4bf h, l;
    float f;
    f = v.x * s; h[0] = (__bf16)f; l[0] = (__bf16)(f - (float)h[0]);
    f = v.y * s; h[1] = (__bf16)f; l[1] = (__bf16)(f - (float)h[1]);
    f = v.z * s; h[2] = (__bf16)f; l[2] = (__bf16)(f - (float)h[2]);
    f = v.w * s; h[3] = (__bf16)f; l[3] = (__bf16)(f - (float)h[3]);
    *reinterpret_cast<v4bf*>(hi) = h;
    *reinterpret_cast<v4bf*>(lo) = l;
}

__global__ __launch_bounds__(128)
void cl_norm_split_kernel(const float* __restrict__ x, const float* __restrict__ y,
                          bf16_t* __restrict__ xhi, bf16_t* __restrict__ xlo,
                          bf16_t* __restrict__ yhi, bf16_t* __restrict__ ylo) {
    const int b   = blockIdx.x;
    const int tid = threadIdx.x;

    const float* src;
    bf16_t *dhi, *dlo;
    int row;
    if (b < B_ROWS) { src = x; dhi = xhi; dlo = xlo; row = b; }
    else            { src = y; dhi = yhi; dlo = ylo; row = b - B_ROWS; }

    const float4* srow = reinterpret_cast<const float4*>(src + (size_t)row * D_DIM);
    float4 v0 = srow[tid];
    float4 v1 = srow[tid + 128];

    float ss = v0.x*v0.x + v0.y*v0.y + v0.z*v0.z + v0.w*v0.w
             + v1.x*v1.x + v1.y*v1.y + v1.z*v1.z + v1.w*v1.w;

    #pragma unroll
    for (int off = 16; off > 0; off >>= 1) ss += __shfl_xor(ss, off, 32);
    __shared__ float wsum[4];
    if ((tid & 31) == 0) wsum[tid >> 5] = ss;
    __syncthreads();
    float tot = wsum[0] + wsum[1] + wsum[2] + wsum[3];

    float scale = 1.0f / fmaxf(sqrtf(tot), EPS_NRM);

    size_t base = (size_t)row * D_DIM;
    cl_split_store(dhi + base + tid * 4,       dlo + base + tid * 4,       v0, scale);
    cl_split_store(dhi + base + tid * 4 + 512, dlo + base + tid * 4 + 512, v1, scale);
}

// ============================================================
// Kernel 2: fused GEMM (bf16x3 split) + row-wise sum(exp) + diag capture
// grid: (B_ROWS/MT, NSEG) x 256 threads (8 waves), wave grid 4(M) x 2(N)
// Double-buffered LDS pipeline; async global->LDS when available.
// ============================================================

// A fragment (16x32 bf16, MxK): lanes 0-15 row M=lane, K {0..7,16..23};
// lanes 16-31 same rows, K {8..15,24..31}
__device__ __forceinline__ v16bf cl_load_frag_a(const bf16_t* slab, int rowBase,
                                                int kh, int lane) {
    int r  = rowBase + (lane & 15);
    int k0 = (lane >> 4) << 3;            // 0 or 8
    const bf16_t* p = slab + r * LDS_STRIDE + kh + k0;
    v8bf a = *reinterpret_cast<const v8bf*>(p);        // K = k0 .. k0+7
    v8bf b = *reinterpret_cast<const v8bf*>(p + 16);   // K = k0+16 .. k0+23
    v16bf f;
    #pragma unroll
    for (int i = 0; i < 8; ++i) { f[i] = a[i]; f[i + 8] = b[i]; }
    return f;
}

// B fragment (32x16 bf16, KxN), column n = row n of yn:
// lanes 0-15 hold K {0..7,16..23} of col n=lane; lanes 16-31 hold K {8..15,24..31}
__device__ __forceinline__ v16bf cl_load_frag_b(const bf16_t* slab, int rowBase,
                                                int kh, int lane) {
    int r  = rowBase + (lane & 15);
    int k0 = (lane >> 4) << 3;            // 0 or 8
    const bf16_t* p = slab + r * LDS_STRIDE + kh + k0;
    v8bf a = *reinterpret_cast<const v8bf*>(p);
    v8bf b = *reinterpret_cast<const v8bf*>(p + 16);
    v16bf f;
    #pragma unroll
    for (int i = 0; i < 8; ++i) { f[i] = a[i]; f[i + 8] = b[i]; }
    return f;
}

// Copy one KSTAGE slice of all 4 slabs into LDS buffer.
// Each thread moves 16 x 16B chunks (4 per slab), no divergent control flow.
__device__ __forceinline__ void cl_issue_copy(const bf16_t* const* g, bf16_t* const* s,
                                              int tid) {
    #pragma unroll
    for (int q = 0; q < 4; ++q) {
        #pragma unroll
        for (int j = 0; j < 4; ++j) {
            const int idx = tid + j * 256;       // 0..1023
            const int rr  = idx >> 3;            // row 0..127
            const int cc  = idx & 7;             // 16B chunk 0..7
            const bf16_t* gp = g[q] + (size_t)rr * D_DIM + cc * 8;
            bf16_t*       sp = s[q] + rr * LDS_STRIDE + cc * 8;
#ifdef USE_ASYNC_LDS
            __builtin_amdgcn_global_load_async_to_lds_b128(AS1PTR(gp), AS3PTR(sp), 0, 0);
#else
            *reinterpret_cast<uint4*>(sp) = *reinterpret_cast<const uint4*>(gp);
#endif
        }
    }
}

__device__ __forceinline__ void cl_wait_copy_prev() {
#ifdef USE_ASYNC_LDS
    __builtin_amdgcn_s_wait_asynccnt(16);   // 16 newer ops may remain in flight
#endif
}
__device__ __forceinline__ void cl_wait_copy_all() {
#ifdef USE_ASYNC_LDS
    __builtin_amdgcn_s_wait_asynccnt(0);
#endif
}

__global__ __launch_bounds__(256)
void cl_gemm_lse_kernel(const bf16_t* __restrict__ xhi, const bf16_t* __restrict__ xlo,
                        const bf16_t* __restrict__ yhi, const bf16_t* __restrict__ ylo,
                        float* __restrict__ rowsum, float* __restrict__ diagv) {
    // [buffer][slab: Ahi,Alo,Bhi,Blo][MT * LDS_STRIDE]
    __shared__ bf16_t sbuf[2][4][MT * LDS_STRIDE];

    const int tid    = threadIdx.x;
    const int lane   = tid & 31;
    const int wid    = tid >> 5;
    const int waveM  = wid & 3;            // 0..3  -> M offset waveM*32
    const int waveN  = wid >> 2;           // 0..1  -> N offset waveN*64
    const int mBlk   = blockIdx.x * MT;
    const int laneLo = lane & 15;
    const int hiHalf = lane >> 4;          // C rows: r + 8*hiHalf

    float pex[2][8];
    #pragma unroll
    for (int mi = 0; mi < 2; ++mi)
        #pragma unroll
        for (int r = 0; r < 8; ++r) pex[mi][r] = 0.0f;

    for (int nt = 0; nt < NT_PER_SEG; ++nt) {
        const int nBase = (blockIdx.y * NT_PER_SEG + nt) * NT;

        v8f acc[2][4];
        #pragma unroll
        for (int mi = 0; mi < 2; ++mi)
            #pragma unroll
            for (int ni = 0; ni < 4; ++ni)
                acc[mi][ni] = v8f{0.f,0.f,0.f,0.f,0.f,0.f,0.f,0.f};

        // ---- pipeline prologue: stage 0 into buffer 0 ----
        {
            const bf16_t* g0[4] = {
                xhi + (size_t)mBlk  * D_DIM,
                xlo + (size_t)mBlk  * D_DIM,
                yhi + (size_t)nBase * D_DIM,
                ylo + (size_t)nBase * D_DIM };
            bf16_t* s0[4] = { sbuf[0][0], sbuf[0][1], sbuf[0][2], sbuf[0][3] };
            cl_issue_copy(g0, s0, tid);
        }

        for (int st = 0; st < NSTAGES; ++st) {
            const int cur = st & 1;
            const int nxt = cur ^ 1;

            __syncthreads();   // all waves done reading buf[nxt] (stage st-1)

            if (st + 1 < NSTAGES) {
                const int kn = (st + 1) * KSTAGE;
                const bf16_t* gn[4] = {
                    xhi + (size_t)mBlk  * D_DIM + kn,
                    xlo + (size_t)mBlk  * D_DIM + kn,
                    yhi + (size_t)nBase * D_DIM + kn,
                    ylo + (size_t)nBase * D_DIM + kn };
                bf16_t* sn[4] = { sbuf[nxt][0], sbuf[nxt][1], sbuf[nxt][2], sbuf[nxt][3] };
                cl_issue_copy(gn, sn, tid);
                cl_wait_copy_prev();   // retire this wave's stage-st copies
            } else {
                cl_wait_copy_all();
            }

            __syncthreads();   // buf[cur] visible to all waves

            const bf16_t* sAhi = sbuf[cur][0];
            const bf16_t* sAlo = sbuf[cur][1];
            const bf16_t* sBhi = sbuf[cur][2];
            const bf16_t* sBlo = sbuf[cur][3];

            #pragma unroll
            for (int kh = 0; kh < KSTAGE; kh += 32) {
                v16bf aH[2], aL[2], bH[4], bL[4];
                #pragma unroll
                for (int mi = 0; mi < 2; ++mi) {
                    aH[mi] = cl_load_frag_a(sAhi, waveM * 32 + mi * 16, kh, lane);
                    aL[mi] = cl_load_frag_a(sAlo, waveM * 32 + mi * 16, kh, lane);
                }
                #pragma unroll
                for (int ni = 0; ni < 4; ++ni) {
                    bH[ni] = cl_load_frag_b(sBhi, waveN * 64 + ni * 16, kh, lane);
                    bL[ni] = cl_load_frag_b(sBlo, waveN * 64 + ni * 16, kh, lane);
                }

                #pragma unroll
                for (int mi = 0; mi < 2; ++mi) {
                    #pragma unroll
                    for (int ni = 0; ni < 4; ++ni) {
                        // bf16x3 split-precision: hi*hi + hi*lo + lo*hi
                        acc[mi][ni] = __builtin_amdgcn_wmma_f32_16x16x32_bf16(
                            false, aH[mi], false, bH[ni], (short)0, acc[mi][ni], false, false);
                        acc[mi][ni] = __builtin_amdgcn_wmma_f32_16x16x32_bf16(
                            false, aH[mi], false, bL[ni], (short)0, acc[mi][ni], false, false);
                        acc[mi][ni] = __builtin_amdgcn_wmma_f32_16x16x32_bf16(
                            false, aL[mi], false, bH[ni], (short)0, acc[mi][ni], false, false);
                    }
                }
            }
        }

        // epilogue: logits bounded by 1/tau -> exp without max-shift is safe in f32
        #pragma unroll
        for (int mi = 0; mi < 2; ++mi) {
            #pragma unroll
            for (int ni = 0; ni < 4; ++ni) {
                const int col = nBase + waveN * 64 + ni * 16 + laneLo;
                #pragma unroll
                for (int r = 0; r < 8; ++r) {
                    float s = acc[mi][ni][r] * INV_TAU;
                    int row = mBlk + waveM * 32 + mi * 16 + hiHalf * 8 + r;
                    if (row == col) diagv[row] = s;   // unique writer
                    pex[mi][r] += __expf(s);
                }
            }
        }
    }

    // reduce exp partials across each 16-lane half (same row group per half)
    #pragma unroll
    for (int mi = 0; mi < 2; ++mi) {
        #pragma unroll
        for (int r = 0; r < 8; ++r) {
            float v = pex[mi][r];
            v += __shfl_xor(v, 1, 32);
            v += __shfl_xor(v, 2, 32);
            v += __shfl_xor(v, 4, 32);
            v += __shfl_xor(v, 8, 32);
            if (laneLo == 0) {
                int row = mBlk + waveM * 32 + mi * 16 + hiHalf * 8 + r;
                atomicAdd(&rowsum[row], v);
            }
        }
    }
}

// ============================================================
// Kernel 3: loss = mean_i( log(rowsum_i) - s_ii )
// ============================================================
__global__ __launch_bounds__(256)
void cl_finalize_kernel(const float* __restrict__ rowsum,
                        const float* __restrict__ diagv,
                        float* __restrict__ out) {
    __shared__ float red[8];
    const int tid = threadIdx.x;
    float s = 0.0f;
    for (int i = tid; i < B_ROWS; i += 256)
        s += logf(rowsum[i]) - diagv[i];
    #pragma unroll
    for (int off = 16; off > 0; off >>= 1) s += __shfl_xor(s, off, 32);
    if ((tid & 31) == 0) red[tid >> 5] = s;
    __syncthreads();
    if (tid == 0) {
        float t = 0.0f;
        #pragma unroll
        for (int i = 0; i < 8; ++i) t += red[i];
        out[0] = t / (float)B_ROWS;
    }
}

// ============================================================
// launch
// ============================================================
extern "C" void kernel_launch(void* const* d_in, const int* in_sizes, int n_in,
                              void* d_out, int out_size, void* d_ws, size_t ws_size,
                              hipStream_t stream) {
    const float* x = (const float*)d_in[0];
    const float* y = (const float*)d_in[1];
    float* out = (float*)d_out;

    const size_t plane = (size_t)B_ROWS * D_DIM;
    bf16_t* xhi = (bf16_t*)d_ws;
    bf16_t* xlo = xhi + plane;
    bf16_t* yhi = xlo + plane;
    bf16_t* ylo = yhi + plane;
    float* rowsum = (float*)(ylo + plane);
    float* diagv  = rowsum + B_ROWS;

    hipLaunchKernelGGL(cl_zero_kernel, dim3(B_ROWS / 256), dim3(256), 0, stream, rowsum);
    hipLaunchKernelGGL(cl_norm_split_kernel, dim3(2 * B_ROWS), dim3(128), 0, stream,
                       x, y, xhi, xlo, yhi, ylo);
    hipLaunchKernelGGL(cl_gemm_lse_kernel, dim3(B_ROWS / MT, NSEG), dim3(256), 0, stream,
                       xhi, xlo, yhi, ylo, rowsum, diagv);
    hipLaunchKernelGGL(cl_finalize_kernel, dim3(1), dim3(256), 0, stream,
                       rowsum, diagv, out);
}
